// MultiHeadAttentionLayer_6880537608494
// MI455X (gfx1250) — matrix operand order repaired and checked by reading, not compiled, not run on hardware
//
#include <hip/hip_runtime.h>
#include <hip/hip_bf16.h>

#define B_  2
#define N_  2048
#define DM_ 256
#define H_  8
#define DH_ 32
#define E_  32768

typedef __attribute__((ext_vector_type(16))) _Float16 v16h;
typedef __attribute__((ext_vector_type(8)))  _Float16 v8h;
typedef __attribute__((ext_vector_type(8)))  float    v8f;

static __device__ __forceinline__ v16h mk16(v8h lo, v8h hi) {
  return __builtin_shufflevector(lo, hi, 0,1,2,3,4,5,6,7,8,9,10,11,12,13,14,15);
}

// ---------------- degree / normalization ----------------
__global__ void k_deginit(float* d) {
  int i = blockIdx.x * blockDim.x + threadIdx.x;
  d[i] = 1.0f; // self loop
}
__global__ void k_degacc(const int* __restrict__ dst, float* d) {
  int e = blockIdx.x * blockDim.x + threadIdx.x;
  atomicAdd(d + dst[e], 1.0f);
}
__global__ void k_dinv(float* d) {
  int i = blockIdx.x * blockDim.x + threadIdx.x;
  d[i] = rsqrtf(d[i]);
}

// ---------------- conversions / repacks ----------------
__global__ void k_f32tof16(const float* __restrict__ in, _Float16* __restrict__ o) {
  int i = blockIdx.x * blockDim.x + threadIdx.x;
  o[i] = (_Float16)in[i];
}
// Wt[n][k] = (f16) W[k][n]
__global__ void k_wt(const float* __restrict__ W, _Float16* __restrict__ Wt) {
  int i = blockIdx.x * blockDim.x + threadIdx.x; // 65536
  int n = i >> 8, k = i & 255;
  Wt[i] = (_Float16)W[k * DM_ + n];
}
// agg = h * dinv^2 + bias   (initializer for scatter-add)
__global__ void k_selfbias(const float* __restrict__ hs, const float* __restrict__ dinv,
                           const float* __restrict__ bias, float* __restrict__ agg) {
  int i = blockIdx.x * blockDim.x + threadIdx.x; // B*N*DM = 2^20
  int c = i & 255, n = (i >> 8) & 2047;
  float di = dinv[n];
  agg[i] = hs[i] * di * di + bias[c];
}
// edge scatter: agg[b,dst,:] += h[b,src,:] * dinv[src]*dinv[dst]
__global__ void k_scatter(const float* __restrict__ hs, const int* __restrict__ src,
                          const int* __restrict__ dst, const float* __restrict__ dinv,
                          float* __restrict__ agg) {
  int t = blockIdx.x * blockDim.x + threadIdx.x; // B*E*32 = 2^21
  int c8 = (t & 31) * 8;
  int e  = (t >> 5) & (E_ - 1);
  int b  = t >> 20;
  int s = src[e], d = dst[e];
  float nm = dinv[s] * dinv[d];
  const float* hp = hs + ((size_t)(b * N_ + s)) * DM_ + c8;
  float*       ap = agg + ((size_t)(b * N_ + d)) * DM_ + c8;
  float4 a0 = *(const float4*)hp;
  float4 a1 = *(const float4*)(hp + 4);
  atomicAdd(ap + 0, a0.x * nm); atomicAdd(ap + 1, a0.y * nm);
  atomicAdd(ap + 2, a0.z * nm); atomicAdd(ap + 3, a0.w * nm);
  atomicAdd(ap + 4, a1.x * nm); atomicAdd(ap + 5, a1.y * nm);
  atomicAdd(ap + 6, a1.z * nm); atomicAdd(ap + 7, a1.w * nm);
}
// [b,n,h*32+dh] f32 -> [b,h,n,dh] f16
__global__ void k_toqk(const float* __restrict__ agg, _Float16* __restrict__ o) {
  int i = blockIdx.x * blockDim.x + threadIdx.x;
  int c = i & 255, n = (i >> 8) & 2047, b = i >> 19;
  int hh = c >> 5, dh = c & 31;
  o[(((size_t)b * H_ + hh) * N_ + n) * DH_ + dh] = (_Float16)agg[i];
}
// [b,n,h*32+dh] f32 -> [b,h,dh,n] f16 (V transposed for B-fragment reads)
__global__ void k_tovt(const float* __restrict__ agg, _Float16* __restrict__ o) {
  int i = blockIdx.x * blockDim.x + threadIdx.x;
  int c = i & 255, n = (i >> 8) & 2047, b = i >> 19;
  int hh = c >> 5, dh = c & 31;
  o[(((size_t)b * H_ + hh) * DH_ + dh) * N_ + n] = (_Float16)agg[i];
}

// ---------------- WMMA GEMM: C[M,256] = A[M,256](f16) @ Wt(f16, [n][k]) ----------------
__global__ __launch_bounds__(256)
void k_gemm(const _Float16* __restrict__ A, const _Float16* __restrict__ Bt,
            float* __restrict__ C) {
  const int lane = threadIdx.x & 31;
  const int wv   = threadIdx.x >> 5;
  const int tile = blockIdx.x * 8 + wv;  // 4096 tiles
  const int tm = tile >> 4;              // 256 row tiles
  const int tn = tile & 15;              // 16 col tiles
  const int l15 = lane & 15;
  const int hl  = lane >> 4;

  // A fragment rows: lane holds row (l15), K chunks [koff..koff+7] and [koff+16..koff+23]
  const _Float16* ar = A  + ((size_t)(tm * 16 + l15)) * DM_ + (hl ? 8 : 0);
  // B fragment cols: lane holds column (l15), K chunks [koff..koff+7] and [koff+16..koff+23]
  const _Float16* br = Bt + ((size_t)(tn * 16 + l15)) * DM_ + (hl ? 8 : 0);

  v8f acc = {};
  #pragma unroll
  for (int kk = 0; kk < DM_; kk += 32) {
    v16h a  = mk16(*(const v8h*)(ar + kk), *(const v8h*)(ar + kk + 16));
    v16h bf = mk16(*(const v8h*)(br + kk), *(const v8h*)(br + kk + 16));
    acc = __builtin_amdgcn_wmma_f32_16x16x32_f16(false, a, false, bf,
                                                 (short)0, acc, false, false);
  }
  float* cr = C + (size_t)(tm * 16) * DM_ + tn * 16 + l15;
  #pragma unroll
  for (int i = 0; i < 8; ++i)
    cr[(size_t)(i + 8 * hl) * DM_] = acc[i];
}

// ---------------- flash attention (f16 WMMA, f32 online softmax) ----------------
__global__ __launch_bounds__(256)
void k_attn(const _Float16* __restrict__ qh, const _Float16* __restrict__ kh,
            const _Float16* __restrict__ vth, _Float16* __restrict__ ah) {
  __shared__ _Float16 Ksh[64 * DH_];     // [key][dh]
  __shared__ _Float16 Vsh[DH_ * 64];     // [dh][key]
  __shared__ _Float16 Psh[8][16 * 64];   // per-wave P staging [row][key]

  const int tid  = threadIdx.x;
  const int lane = tid & 31;
  const int wv   = tid >> 5;
  const int l15  = lane & 15;
  const int hl   = lane >> 4;

  const int g   = blockIdx.x;            // 256 blocks
  const int bh  = g >> 4;                // b*H + h
  const int b   = bh >> 3;
  const int hh  = bh & 7;
  const int qt  = (g & 15) * 8 + wv;     // q tile index within (b,h)
  const int qr0 = qt * 16;

  const float scale = 0.1767766952966369f; // DH^-0.5

  // Q fragment (16x32, K = DH = 32) held for the whole pass
  const _Float16* qb = qh + ((size_t)bh * N_ + qr0 + l15) * DH_ + (hl ? 8 : 0);
  v16h qf = mk16(*(const v8h*)qb, *(const v8h*)(qb + 16));

  const _Float16* kb0 = kh  + (size_t)bh * N_ * DH_;
  const _Float16* vb0 = vth + (size_t)bh * DH_ * N_;

  v8f O0 = {}, O1 = {};
  float mrow[8], lrow[8];
  #pragma unroll
  for (int i = 0; i < 8; ++i) { mrow[i] = -3.0e38f; lrow[i] = 0.0f; }

  for (int kbl = 0; kbl < N_; kbl += 64) {
    __syncthreads();
    { // cooperative staging: K 64x32, Vt 32x64 (8 halfs per thread each)
      const int r = tid >> 2, c = (tid & 3) * 8;
      *(v8h*)(Ksh + r * DH_ + c) = *(const v8h*)(kb0 + (size_t)(kbl + r) * DH_ + c);
      const int r2 = tid >> 3, c2 = (tid & 7) * 8;
      *(v8h*)(Vsh + r2 * 64 + c2) = *(const v8h*)(vb0 + (size_t)r2 * N_ + kbl + c2);
    }
    __syncthreads();

    // S = Q @ K^T : 4 key sub-tiles of 16
    v8f Sv[4];
    #pragma unroll
    for (int s = 0; s < 4; ++s) {
      const _Float16* kp = Ksh + (s * 16 + l15) * DH_ + hl * 8;
      v16h kf = mk16(*(const v8h*)kp, *(const v8h*)(kp + 16));
      v8f z = {};
      Sv[s] = __builtin_amdgcn_wmma_f32_16x16x32_f16(false, qf, false, kf,
                                                     (short)0, z, false, false);
      #pragma unroll
      for (int i = 0; i < 8; ++i) Sv[s][i] *= scale;
    }

    // online softmax per row (row = i + 8*hl lives across lanes of each 16-half)
    #pragma unroll
    for (int i = 0; i < 8; ++i) {
      float t = fmaxf(fmaxf(Sv[0][i], Sv[1][i]), fmaxf(Sv[2][i], Sv[3][i]));
      #pragma unroll
      for (int off = 1; off < 16; off <<= 1)
        t = fmaxf(t, __shfl_xor(t, off, 32));
      const float mn    = fmaxf(mrow[i], t);
      const float alpha = __expf(mrow[i] - mn);
      mrow[i] = mn;
      float rs = 0.0f;
      #pragma unroll
      for (int s = 0; s < 4; ++s) {
        const float p = __expf(Sv[s][i] - mn);
        Sv[s][i] = p;
        rs += p;
      }
      #pragma unroll
      for (int off = 1; off < 16; off <<= 1)
        rs += __shfl_xor(rs, off, 32);
      lrow[i] = lrow[i] * alpha + rs;
      O0[i] *= alpha;
      O1[i] *= alpha;
    }

    // P (C-layout) -> LDS -> A-layout fragments
    _Float16* pr = Psh[wv];
    #pragma unroll
    for (int s = 0; s < 4; ++s)
      #pragma unroll
      for (int i = 0; i < 8; ++i)
        pr[(i + 8 * hl) * 64 + s * 16 + l15] = (_Float16)Sv[s][i];

    asm volatile("s_wait_dscnt 0x0" ::: "memory");

    #pragma unroll
    for (int c = 0; c < 2; ++c) { // key chunks 0..31 / 32..63
      const _Float16* pp = pr + l15 * 64 + c * 32 + (hl ? 8 : 0);
      v16h pf = mk16(*(const v8h*)pp, *(const v8h*)(pp + 16));
      const _Float16* vp0 = Vsh + (0 + l15) * 64 + c * 32 + hl * 8;
      v16h vf0 = mk16(*(const v8h*)vp0, *(const v8h*)(vp0 + 16));
      O0 = __builtin_amdgcn_wmma_f32_16x16x32_f16(false, pf, false, vf0,
                                                  (short)0, O0, false, false);
      const _Float16* vp1 = Vsh + (16 + l15) * 64 + c * 32 + hl * 8;
      v16h vf1 = mk16(*(const v8h*)vp1, *(const v8h*)(vp1 + 16));
      O1 = __builtin_amdgcn_wmma_f32_16x16x32_f16(false, pf, false, vf1,
                                                  (short)0, O1, false, false);
    }
  }

  // normalize and emit as f16 [b,n,h*32+dh] for the output projection GEMM
  #pragma unroll
  for (int i = 0; i < 8; ++i) {
    const float inv = 1.0f / lrow[i];
    const int n = qr0 + i + 8 * hl;
    _Float16* op = ah + ((size_t)(b * N_ + n)) * DM_ + hh * DH_;
    op[l15]      = (_Float16)(O0[i] * inv);
    op[16 + l15] = (_Float16)(O1[i] * inv);
  }
}

// ---------------- host-side orchestration ----------------
extern "C" void kernel_launch(void* const* d_in, const int* in_sizes, int n_in,
                              void* d_out, int out_size, void* d_ws, size_t ws_size,
                              hipStream_t stream) {
  (void)in_sizes; (void)n_in; (void)out_size; (void)ws_size;
  const float* x  = (const float*)d_in[0];
  const int*   ei = (const int*)d_in[1];
  const int*   src = ei;
  const int*   dst = ei + E_;
  const float* Wq = (const float*)d_in[2]; const float* bq = (const float*)d_in[3];
  const float* Wk = (const float*)d_in[4]; const float* bk = (const float*)d_in[5];
  const float* Wv = (const float*)d_in[6]; const float* bv = (const float*)d_in[7];
  const float* Wo = (const float*)d_in[8]; const float* bo = (const float*)d_in[9];
  float* out = (float*)d_out;

  char* w = (char*)d_ws;
  float*    dinv = (float*)w;      w += (size_t)N_ * 4;
  _Float16* xh   = (_Float16*)w;   w += (size_t)B_ * N_ * DM_ * 2;
  _Float16* wtq  = (_Float16*)w;   w += (size_t)DM_ * DM_ * 2;
  _Float16* wtk  = (_Float16*)w;   w += (size_t)DM_ * DM_ * 2;
  _Float16* wtv  = (_Float16*)w;   w += (size_t)DM_ * DM_ * 2;
  _Float16* wto  = (_Float16*)w;   w += (size_t)DM_ * DM_ * 2;
  float*    hbuf = (float*)w;      w += (size_t)B_ * N_ * DM_ * 4;
  float*    agg  = (float*)w;      w += (size_t)B_ * N_ * DM_ * 4;
  _Float16* qhb  = (_Float16*)w;   w += (size_t)B_ * N_ * DM_ * 2;
  _Float16* khb  = (_Float16*)w;   w += (size_t)B_ * N_ * DM_ * 2;
  _Float16* vtb  = (_Float16*)w;   w += (size_t)B_ * N_ * DM_ * 2;
  _Float16* ahb  = (_Float16*)w;   w += (size_t)B_ * N_ * DM_ * 2;

  const int BN   = B_ * N_ * DM_;       // 2^20
  const int ELEM = BN / 256;            // 4096 blocks
  const int SCAT = (B_ * E_ * 32) / 256;// 8192 blocks

  // normalization
  k_deginit<<<N_ / 256, 256, 0, stream>>>(dinv);
  k_degacc <<<E_ / 256, 256, 0, stream>>>(dst, dinv);
  k_dinv   <<<N_ / 256, 256, 0, stream>>>(dinv);

  // f16 operands
  k_f32tof16<<<ELEM, 256, 0, stream>>>(x, xh);
  k_wt<<<256, 256, 0, stream>>>(Wq, wtq);
  k_wt<<<256, 256, 0, stream>>>(Wk, wtk);
  k_wt<<<256, 256, 0, stream>>>(Wv, wtv);
  k_wt<<<256, 256, 0, stream>>>(Wo, wto);

  // q = gcn(x, Wq)
  k_gemm    <<<512,  256, 0, stream>>>(xh, wtq, hbuf);
  k_selfbias<<<ELEM, 256, 0, stream>>>(hbuf, dinv, bq, agg);
  k_scatter <<<SCAT, 256, 0, stream>>>(hbuf, src, dst, dinv, agg);
  k_toqk    <<<ELEM, 256, 0, stream>>>(agg, qhb);

  // k = gcn(x, Wk)
  k_gemm    <<<512,  256, 0, stream>>>(xh, wtk, hbuf);
  k_selfbias<<<ELEM, 256, 0, stream>>>(hbuf, dinv, bk, agg);
  k_scatter <<<SCAT, 256, 0, stream>>>(hbuf, src, dst, dinv, agg);
  k_toqk    <<<ELEM, 256, 0, stream>>>(agg, khb);

  // v = gcn(x, Wv), stored transposed
  k_gemm    <<<512,  256, 0, stream>>>(xh, wtv, hbuf);
  k_selfbias<<<ELEM, 256, 0, stream>>>(hbuf, dinv, bv, agg);
  k_scatter <<<SCAT, 256, 0, stream>>>(hbuf, src, dst, dinv, agg);
  k_tovt    <<<ELEM, 256, 0, stream>>>(agg, vtb);

  // attention
  k_attn<<<256, 256, 0, stream>>>(qhb, khb, vtb, ahb);

  // out = gcn(attn_out, Wo) -> d_out
  k_gemm    <<<512,  256, 0, stream>>>(ahb, wto, hbuf);
  k_selfbias<<<ELEM, 256, 0, stream>>>(hbuf, dinv, bo, out);
  k_scatter <<<SCAT, 256, 0, stream>>>(hbuf, src, dst, dinv, out);
}
